// MultiheadAttention_23261542875740
// MI455X (gfx1250) — matrix operand order repaired and checked by reading, not compiled, not run on hardware
//
#include <hip/hip_runtime.h>

// ---------------------------------------------------------------------------
// MultiheadAttention for MI455X (gfx1250): f16 WMMA pipeline, f32 accumulate.
//   B=4, S=2048, NH=16, DIN=1024, HD=64, rows = B*S = 8192
// All three GEMM kernels stream their shared B operand through double-buffered
// LDS using GLOBAL_LOAD_ASYNC_TO_LDS_B128 (ASYNCcnt), so WMMAs consume
// low-latency ds_load fragments while the next chunk's DMA is in flight.
// ---------------------------------------------------------------------------

#define NH   16
#define BB   4
#define SS   2048
#define DIN  1024
#define HD   64
#define ROWS (BB * SS)          // 8192
#define INV_SQRT_D 0.125f       // 1/sqrt(64)

typedef __attribute__((ext_vector_type(16))) _Float16 v16h;
typedef __attribute__((ext_vector_type(8)))  float    v8f;
typedef __attribute__((ext_vector_type(8)))  unsigned v8u;
typedef __attribute__((ext_vector_type(4)))  _Float16 v4h;

union FragU { v8u u; v16h h; };

// A-matrix 16x32 f16 fragment (ISA 7.12.2): lanes 0-15 -> M=lane, K {0..7,16..23};
// lanes 16-31 -> M=lane-16, K {8..15,24..31}. Two packed halfs per VGPR.
__device__ __forceinline__ v16h load_frag_a(const _Float16* base, int ld) {
  const int lane = threadIdx.x & 31;
  const int m    = lane & 15;
  const int koff = (lane >> 4) << 3;        // 0 or 8
  const _Float16* p = base + m * ld + koff;
  FragU f;
#pragma unroll
  for (int v = 0; v < 4; ++v) {
    f.u[v]     = *(const unsigned*)(p + 2 * v);        // K = koff + 2v .. +1
    f.u[v + 4] = *(const unsigned*)(p + 16 + 2 * v);   // K = 16 + koff + 2v
  }
  return f.h;
}

// B-matrix 32x16 f16 fragment: lanes 0-15 -> N=lane, K {0..7,16..23}; lanes 16-31 -> K {8..15,24..31}.
// Memory must be N-major (i.e. B^T row-major) so K is contiguous.
__device__ __forceinline__ v16h load_frag_b(const _Float16* base, int ld) {
  const int lane = threadIdx.x & 31;
  const int n    = lane & 15;
  const int koff = (lane >> 4) << 3;        // 0 or 8
  const _Float16* p = base + n * ld + koff;
  FragU f;
#pragma unroll
  for (int v = 0; v < 8; ++v)
    f.u[v] = *(const unsigned*)(p + 2 * v + (v >= 4 ? 8 : 0));
  return f.h;
}

__device__ __forceinline__ v8f wmma32(v16h a, v16h b, v8f c) {
  // D = A(16x32) * B(32x16) + C, f32 accumulate
  return __builtin_amdgcn_wmma_f32_16x16x32_f16(false, a, false, b, (short)0, c,
                                                false, false);
}

// ---- CDNA5 async global->LDS (ASYNCcnt) -----------------------------------
// VDST supplies the per-lane LDS byte address; generic shared addr[31:0] is the
// LDS offset (ISA 10.2 aperture mapping).
__device__ __forceinline__ void async_b128(const void* gsrc, void* ldst) {
  unsigned loff = (unsigned)(size_t)ldst;
  asm volatile("global_load_async_to_lds_b128 %0, %1, off"
               :: "v"(loff), "v"(gsrc) : "memory");
}
#define WAIT_ASYNC(n) asm volatile("s_wait_asynccnt %0" :: "i"(n) : "memory")

// ---------------------------------------------------------------------------
// 1) q f32 -> f16
// ---------------------------------------------------------------------------
__global__ void cvt_f32_f16_kernel(const float* __restrict__ s,
                                   _Float16* __restrict__ d, int n) {
  const int i      = blockIdx.x * blockDim.x + threadIdx.x;
  const int stride = gridDim.x * blockDim.x;
  const float4* s4 = (const float4*)s;
  v4h* d4 = (v4h*)d;
  for (int k = i; k < (n >> 2); k += stride) {
    float4 v = s4[k];
    v4h o;
    o.x = (_Float16)v.x; o.y = (_Float16)v.y;
    o.z = (_Float16)v.z; o.w = (_Float16)v.w;
    d4[k] = o;
  }
}

// ---------------------------------------------------------------------------
// 2) Pack W (NH,DIN,HD) f32 -> WT[(headBase+h)*64 + n][k] f16   (N-major)
// ---------------------------------------------------------------------------
__global__ void pack_w_kernel(const float* __restrict__ W,
                              _Float16* __restrict__ WT, int headBase) {
  const int idx = blockIdx.x * blockDim.x + threadIdx.x;   // [0, 1M)
  const int k = idx & (DIN - 1);
  const int n = (idx >> 10) & (HD - 1);
  const int h = (idx >> 16) & (NH - 1);
  WT[(size_t)((headBase + h) * HD + n) * DIN + k] =
      (_Float16)W[(size_t)(h * DIN + k) * HD + n];
}

// W_out (NH,HD,EMBED) f32 -> WoT[e][h*64+v] f16  (E-major, K = h*64+v contiguous)
__global__ void pack_wo_kernel(const float* __restrict__ Wo,
                               _Float16* __restrict__ WoT) {
  const int idx = blockIdx.x * blockDim.x + threadIdx.x;   // [0, 1M)
  const int kk = idx & (DIN - 1);
  const int e  = idx >> 10;
  WoT[(size_t)e * DIN + kk] = (_Float16)Wo[(size_t)kk * DIN + e];
}

// ---------------------------------------------------------------------------
// 3) Q/K/V projections. Grid: 16 heads x 128 blocks, 4 waves/block; all four
//    waves share head h, so the 12 KB weight chunk of each k-step (3 x 64 rows
//    x 32 halfs) is staged once into double-buffered LDS with async DMA
//    (6 b128/thread/chunk, s_wait_asynccnt 6 pipelined against the WMMAs).
//    B fragments then come from ds_load; the A fragment is register
//    double-buffered so its global latency hides under the 12 WMMAs.
//    Qh,Kh: [h][row][d] f16.  VT: [h][b][d][s] f16 (transposed for P*V).
// ---------------------------------------------------------------------------
__global__ void __launch_bounds__(128)
qkv_proj_kernel(const _Float16* __restrict__ qh,
                const _Float16* __restrict__ WT,   // [3*16][64][1024]
                _Float16* __restrict__ Qh,
                _Float16* __restrict__ Kh,
                _Float16* __restrict__ VT) {
  __shared__ _Float16 Wc[2][3 * HD * 32];   // 2 x 12 KB  [w*64+n][k 0..31]

  const int tid   = threadIdx.x;
  const int wave  = tid >> 5;
  const int lane  = tid & 31;
  const int h     = blockIdx.x >> 7;                       // 0..15
  const int mt    = ((blockIdx.x & 127) << 2) + wave;      // 0..511
  const int mbase = mt * 16;

  // stage one 32-wide weight chunk (Q,K,V panels for head h) into buffer p
  auto stage = [&](int kk, int p) {
#pragma unroll
    for (int q = 0; q < 6; ++q) {
      const int s   = tid + q * 128;        // slice 0..767 (16B each)
      const int w   = s >> 8;               // 0..2
      const int r   = (s & 255) >> 2;       // 0..63
      const int prt = (s & 3) * 8;          // halfs within row
      async_b128(WT + (size_t)((w * NH + h) * HD + r) * DIN + kk + prt,
                 &Wc[p][(w * HD + r) * 32 + prt]);
    }
  };

  v8f acc[3][4];
#pragma unroll
  for (int w = 0; w < 3; ++w)
#pragma unroll
    for (int nt = 0; nt < 4; ++nt) acc[w][nt] = (v8f){};

  const _Float16* abase = qh + (size_t)mbase * DIN;

  stage(0, 0);
  v16h acur = load_frag_a(abase, DIN);

#pragma unroll 1
  for (int kt = 0; kt < DIN / 32; ++kt) {
    const int p = kt & 1;
    __syncthreads();                        // buffer p^1 free to overwrite
    if (kt + 1 < DIN / 32) {
      stage((kt + 1) * 32, p ^ 1);          // next chunk DMA in flight
      WAIT_ASYNC(6);                        // chunk kt landed (in-order)
    } else {
      WAIT_ASYNC(0);
    }
    __syncthreads();                        // all waves' slices visible

    v16h anext = acur;
    if (kt + 1 < DIN / 32) {
      __builtin_prefetch(abase + (kt + 2) * 32, 0, 0);
      anext = load_frag_a(abase + (kt + 1) * 32, DIN);     // overlaps WMMAs
    }
#pragma unroll
    for (int w = 0; w < 3; ++w)
#pragma unroll
      for (int nt = 0; nt < 4; ++nt) {
        const v16h bfr = load_frag_b(&Wc[p][(w * HD + nt * 16) * 32], 32);
        acc[w][nt] = wmma32(acur, bfr, acc[w][nt]);
      }
    acur = anext;
  }

  // C/D layout: VGPR r -> M = r + 8*(lane>>4); N = lane&15 (+16*nt)
  const int n0 = lane & 15;
  const int rh = (lane >> 4) * 8;
#pragma unroll
  for (int r = 0; r < 8; ++r) {
    const int m  = mbase + rh + r;          // global row in [0, 8192)
    const int b_ = m >> 11;                 // batch
    const int sl = m & (SS - 1);            // seq pos within batch
#pragma unroll
    for (int nt = 0; nt < 4; ++nt) {
      const int d = nt * 16 + n0;
      Qh[(size_t)(h * ROWS + m) * HD + d] = (_Float16)acc[0][nt][r];
      Kh[(size_t)(h * ROWS + m) * HD + d] = (_Float16)acc[1][nt][r];
      VT[(size_t)((h * BB + b_) * HD + d) * SS + sl] = (_Float16)acc[2][nt][r];
    }
  }
}

// ---------------------------------------------------------------------------
// 4) Flash attention. Grid: 16h x 4b x 32 blocks, 4 waves/block; the 4 waves
//    share one (h,b) K/V stream: each 32-wide chunk (4KB K + 4KB V) is staged
//    into double-buffered LDS with async DMA (4 b128/thread/chunk), next
//    chunk's DMA overlapping current chunk's WMMAs. Output H: [b][s][h*64+d].
// ---------------------------------------------------------------------------
__global__ void __launch_bounds__(128)
flash_attn_kernel(const _Float16* __restrict__ Qh,
                  const _Float16* __restrict__ Kh,
                  const _Float16* __restrict__ VT,
                  _Float16* __restrict__ H) {
  __shared__ _Float16 Kbuf[2][32 * HD];     // 8 KB  [s 0..31][d 0..63]
  __shared__ _Float16 Vbuf[2][HD * 32];     // 8 KB  [d 0..63][s 0..31]
  __shared__ _Float16 pbuf[4][16][32];      // 4 KB  per-wave P staging

  const int tid  = threadIdx.x;
  const int wave = tid >> 5;
  const int lane = tid & 31;
  const int h    = blockIdx.x >> 7;                        // 0..15
  const int rem  = blockIdx.x & 127;
  const int b    = rem >> 5;                               // 0..3
  const int qt   = ((rem & 31) << 2) + wave;               // 0..127
  const int qbase = qt * 16;

  const _Float16* Qb = Qh + (size_t)(h * ROWS + b * SS + qbase) * HD;
  const _Float16* Kb = Kh + (size_t)(h * ROWS + b * SS) * HD;
  const _Float16* Vb = VT + (size_t)(h * BB + b) * HD * SS;

  auto stage = [&](int j, int p) {
    // K chunk: rows j..j+31 x 64 halfs, fully contiguous (ld == HD): 256 slices
    const char* kg = (const char*)(Kb + (size_t)j * HD);
    char* kl = (char*)&Kbuf[p][0];
    async_b128(kg + tid * 16, kl + tid * 16);
    async_b128(kg + (tid + 128) * 16, kl + (tid + 128) * 16);
    // V chunk: 64 d-rows x 32 halfs at col j (ld == SS): 4 slices of 8h per row
#pragma unroll
    for (int q = 0; q < 2; ++q) {
      const int s   = tid + q * 128;        // slice 0..255
      const int d   = s >> 2;
      const int prt = (s & 3) * 8;
      async_b128(Vb + (size_t)d * SS + j + prt, &Vbuf[p][d * 32 + prt]);
    }
  };

  // Q fragments, resident for the whole row: K-dim = 64 -> two 16x32 A frags
  const v16h qa0 = load_frag_a(Qb, HD);
  const v16h qa1 = load_frag_a(Qb + 32, HD);

  float mrow[8], lrow[8], alpha[8];
  v8f o[4];
#pragma unroll
  for (int r = 0; r < 8; ++r) { mrow[r] = -3.0e38f; lrow[r] = 0.0f; }
#pragma unroll
  for (int nt = 0; nt < 4; ++nt) o[nt] = (v8f){};

  const int c  = lane & 15;
  const int rh = (lane >> 4) * 8;

  stage(0, 0);                              // prologue: chunk 0 -> buffer 0

#pragma unroll 1
  for (int jt = 0; jt < SS / 32; ++jt) {
    const int p = jt & 1;
    __syncthreads();                        // buffer p^1 free to overwrite
    if (jt + 1 < SS / 32) {
      stage((jt + 1) * 32, p ^ 1);          // prefetch next chunk (in flight)
      WAIT_ASYNC(4);                        // chunk jt landed (in-order)
    } else {
      WAIT_ASYNC(0);
    }
    __syncthreads();                        // all waves' slices of jt visible

    const _Float16* Kc = &Kbuf[p][0];
    const _Float16* Vc = &Vbuf[p][0];

    // ---- scores S[16 x 32] = Q * K^T (ds_load fragments from LDS) ----------
    v8f s0 = (v8f){}, s1 = (v8f){};
    s0 = wmma32(qa0, load_frag_b(Kc, HD), s0);
    s0 = wmma32(qa1, load_frag_b(Kc + 32, HD), s0);
    s1 = wmma32(qa0, load_frag_b(Kc + 16 * HD, HD), s1);
    s1 = wmma32(qa1, load_frag_b(Kc + 16 * HD + 32, HD), s1);

    // ---- online softmax (row reductions across the 16-lane column group) ---
#pragma unroll
    for (int r = 0; r < 8; ++r) {
      float a0 = s0[r] * INV_SQRT_D;
      float a1 = s1[r] * INV_SQRT_D;
      float mx = fmaxf(a0, a1);
      mx = fmaxf(mx, __shfl_xor(mx, 1, 32));
      mx = fmaxf(mx, __shfl_xor(mx, 2, 32));
      mx = fmaxf(mx, __shfl_xor(mx, 4, 32));
      mx = fmaxf(mx, __shfl_xor(mx, 8, 32));
      const float mnew = fmaxf(mrow[r], mx);
      alpha[r] = __expf(mrow[r] - mnew);
      mrow[r]  = mnew;
      const float p0 = __expf(a0 - mnew);
      const float p1 = __expf(a1 - mnew);
      s0[r] = p0; s1[r] = p1;
      float ps = p0 + p1;
      ps += __shfl_xor(ps, 1, 32);
      ps += __shfl_xor(ps, 2, 32);
      ps += __shfl_xor(ps, 4, 32);
      ps += __shfl_xor(ps, 8, 32);
      lrow[r] = lrow[r] * alpha[r] + ps;
    }

    // ---- P register-layout (C-frag) -> A-frag via LDS ----------------------
#pragma unroll
    for (int r = 0; r < 8; ++r) {
      pbuf[wave][rh + r][c]      = (_Float16)s0[r];
      pbuf[wave][rh + r][16 + c] = (_Float16)s1[r];
    }
    const v16h pa = load_frag_a(&pbuf[wave][0][0], 32);

    // ---- O = O*alpha + P * V ----------------------------------------------
#pragma unroll
    for (int nt = 0; nt < 4; ++nt) {
#pragma unroll
      for (int r = 0; r < 8; ++r) o[nt][r] *= alpha[r];
      const v16h vb = load_frag_b(Vc + nt * 16 * 32, 32);
      o[nt] = wmma32(pa, vb, o[nt]);
    }
  }

  // ---- epilogue: O /= l, store heads as [b][s][h*64+d] ---------------------
#pragma unroll
  for (int r = 0; r < 8; ++r) {
    const float inv_l = 1.0f / lrow[r];
    const size_t row = (size_t)(b * SS + qbase + rh + r) * DIN + h * HD;
#pragma unroll
    for (int nt = 0; nt < 4; ++nt)
      H[row + nt * 16 + c] = (_Float16)(o[nt][r] * inv_l);
  }
}

// ---------------------------------------------------------------------------
// 5) Output projection: out[8192,1024] = H[8192,1024] * WoT^T  (f32 out)
//    The 4 waves of a block share one 64-column weight panel; its 4 KB
//    k-chunk is staged async into double-buffered LDS (2 b128/thread/chunk);
//    A is register double-buffered. Grid: 2048 blocks x 128 threads.
// ---------------------------------------------------------------------------
__global__ void __launch_bounds__(128)
out_proj_kernel(const _Float16* __restrict__ H,
                const _Float16* __restrict__ WoT,
                float* __restrict__ out) {
  __shared__ _Float16 Wc[2][HD * 32];       // 2 x 4 KB  [n 0..63][k 0..31]

  const int tid  = threadIdx.x;
  const int wave = tid >> 5;
  const int lane = tid & 31;
  const int ng   = blockIdx.x & 15;                        // shared by 4 waves
  const int mt   = ((blockIdx.x >> 4) << 2) + wave;        // 0..511
  const int mbase = mt * 16;
  const int nbase = ng * 64;

  auto stage = [&](int kk, int p) {
#pragma unroll
    for (int q = 0; q < 2; ++q) {
      const int s   = tid + q * 128;        // slice 0..255
      const int r   = s >> 2;               // 0..63
      const int prt = (s & 3) * 8;
      async_b128(WoT + (size_t)(nbase + r) * DIN + kk + prt,
                 &Wc[p][r * 32 + prt]);
    }
  };

  v8f acc[4];
#pragma unroll
  for (int nt = 0; nt < 4; ++nt) acc[nt] = (v8f){};

  const _Float16* abase = H + (size_t)mbase * DIN;

  stage(0, 0);
  v16h acur = load_frag_a(abase, DIN);

#pragma unroll 1
  for (int kt = 0; kt < DIN / 32; ++kt) {
    const int p = kt & 1;
    __syncthreads();
    if (kt + 1 < DIN / 32) {
      stage((kt + 1) * 32, p ^ 1);
      WAIT_ASYNC(2);
    } else {
      WAIT_ASYNC(0);
    }
    __syncthreads();

    v16h anext = acur;
    if (kt + 1 < DIN / 32) {
      __builtin_prefetch(abase + (kt + 2) * 32, 0, 0);
      anext = load_frag_a(abase + (kt + 1) * 32, DIN);
    }
#pragma unroll
    for (int nt = 0; nt < 4; ++nt) {
      const v16h bfr = load_frag_b(&Wc[p][nt * 16 * 32], 32);
      acc[nt] = wmma32(acur, bfr, acc[nt]);
    }
    acur = anext;
  }

  const int n0 = lane & 15;
  const int rh = (lane >> 4) * 8;
#pragma unroll
  for (int r = 0; r < 8; ++r) {
    const size_t row = (size_t)(mbase + rh + r) * DIN;
#pragma unroll
    for (int nt = 0; nt < 4; ++nt)
      out[row + nbase + nt * 16 + n0] = acc[nt][r];
  }
}

// ---------------------------------------------------------------------------
// Host launcher
// ---------------------------------------------------------------------------
extern "C" void kernel_launch(void* const* d_in, const int* in_sizes, int n_in,
                              void* d_out, int out_size, void* d_ws,
                              size_t ws_size, hipStream_t stream) {
  (void)in_sizes; (void)n_in; (void)out_size; (void)ws_size;
  const float* q  = (const float*)d_in[0];
  const float* Wq = (const float*)d_in[1];
  const float* Wk = (const float*)d_in[2];
  const float* Wv = (const float*)d_in[3];
  const float* Wo = (const float*)d_in[4];
  float* out = (float*)d_out;

  char* ws = (char*)d_ws;
  size_t off = 0;
  auto alloc = [&](size_t bytes) -> void* {
    void* p = ws + off;
    off += (bytes + 255) & ~(size_t)255;
    return p;
  };

  _Float16* qh  = (_Float16*)alloc((size_t)ROWS * DIN * 2);          // 16 MB
  _Float16* WT  = (_Float16*)alloc((size_t)3 * NH * HD * DIN * 2);   //  6 MB
  _Float16* WoT = (_Float16*)alloc((size_t)DIN * DIN * 2);           //  2 MB
  _Float16* Qh  = (_Float16*)alloc((size_t)NH * ROWS * HD * 2);      // 16 MB
  _Float16* Kh  = (_Float16*)alloc((size_t)NH * ROWS * HD * 2);      // 16 MB
  _Float16* VT  = (_Float16*)alloc((size_t)NH * ROWS * HD * 2);      // 16 MB
  _Float16* Hh  = (_Float16*)alloc((size_t)ROWS * DIN * 2);          // 16 MB

  cvt_f32_f16_kernel<<<4096, 256, 0, stream>>>(q, qh, ROWS * DIN);
  pack_w_kernel<<<4096, 256, 0, stream>>>(Wq, WT, 0);
  pack_w_kernel<<<4096, 256, 0, stream>>>(Wk, WT, NH);
  pack_w_kernel<<<4096, 256, 0, stream>>>(Wv, WT, 2 * NH);
  pack_wo_kernel<<<4096, 256, 0, stream>>>(Wo, WoT);

  qkv_proj_kernel<<<NH * 128, 128, 0, stream>>>(qh, WT, Qh, Kh, VT);
  flash_attn_kernel<<<NH * BB * 32, 128, 0, stream>>>(Qh, Kh, VT, Hh);
  out_proj_kernel<<<2048, 128, 0, stream>>>(Hh, WoT, out);
}
